// PolymerGNN_30648886624267
// MI455X (gfx1250) — matrix operand order repaired and checked by reading, not compiled, not run on hardware
//
#include <hip/hip_runtime.h>
#include <hip/hip_bf16.h>

typedef __attribute__((ext_vector_type(16))) _Float16 v16h;
typedef __attribute__((ext_vector_type(8)))  _Float16 v8h;
typedef __attribute__((ext_vector_type(8)))  float    v8f;

#define HID 128
#define HEADS 4
#define HDIM 32

// ---------- helpers ----------
__device__ __forceinline__ unsigned f32_ord(float f) {
    unsigned u = __float_as_uint(f);
    return (u & 0x80000000u) ? ~u : (u | 0x80000000u);
}
__device__ __forceinline__ float ord_f32(unsigned u) {
    return __uint_as_float((u & 0x80000000u) ? (u & 0x7FFFFFFFu) : ~u);
}
__device__ __forceinline__ float leaky02(float v) { return v > 0.0f ? v : 0.2f * v; }

union frag16 { v16h v; v8h h[2]; };

// ---------- generic small kernels ----------
__global__ void k_f32_to_f16(const float* __restrict__ s, _Float16* __restrict__ d, int n) {
    int i = blockIdx.x * blockDim.x + threadIdx.x;
    if (i < n) d[i] = (_Float16)s[i];
}
// W [K,128] row-major f32  ->  Wt [128,K] f16 (column of W contiguous)
__global__ void k_w_to_f16_t(const float* __restrict__ W, _Float16* __restrict__ Wt, int K) {
    int idx = blockIdx.x * blockDim.x + threadIdx.x;
    if (idx >= K * HID) return;
    int k = idx / HID, c = idx % HID;
    Wt[c * K + k] = (_Float16)W[k * HID + c];
}
__global__ void k_fill1(float* __restrict__ p, int n) {
    int i = blockIdx.x * blockDim.x + threadIdx.x;
    if (i < n) p[i] = 1.0f;
}
__global__ void k_deg_edges(const int* __restrict__ dst, float* __restrict__ deg, int E) {
    int e = blockIdx.x * blockDim.x + threadIdx.x;
    if (e < E) atomicAdd(&deg[dst[e]], 1.0f);
}
__global__ void k_rsqrt_inplace(float* __restrict__ p, int n) {
    int i = blockIdx.x * blockDim.x + threadIdx.x;
    if (i < n) { float v = p[i]; p[i] = (v > 0.0f) ? rsqrtf(v) : 0.0f; }
}

// ---------- WMMA GEMM: C[N,128] = A[N,K](f16,row) * Bt[128,K](f16,col-of-B contiguous) ----------
// One wave computes a 16x16 tile of C. 8 waves per block => 16 rows x 128 cols.
template <int K>
__global__ __launch_bounds__(256)
void k_gemm_wmma(const _Float16* __restrict__ Ah, const _Float16* __restrict__ Bt,
                 float* __restrict__ C, int N) {
    const int wave = threadIdx.x >> 5;            // 0..7 -> column tile
    const int lane = threadIdx.x & 31;
    const int m0   = blockIdx.x * 16;
    const int n0   = wave * 16;
    const int r16  = lane & 15;
    const int hs   = lane >> 4;                   // half-select
    const int row  = min(m0 + r16, N - 1);        // clamp (reads only; stores guarded)
    const int col  = n0 + r16;

    const _Float16* __restrict__ Arow = Ah + (size_t)row * K;
    const _Float16* __restrict__ Bcol = Bt + (size_t)col * K;

    v8f c = {};
    if constexpr (K >= 32) {
#pragma unroll
        for (int kb = 0; kb < K; kb += 32) {
            frag16 a, b;
            // A 16x32: lanes 0-15 hold K=kb+0..7 & kb+16..23; lanes 16-31 shifted by 8
            a.h[0] = *(const v8h*)&Arow[kb + hs * 8];
            a.h[1] = *(const v8h*)&Arow[kb + 16 + hs * 8];
            // B 32x16: lanes 0-15 hold K=kb+0..7 & kb+16..23; lanes 16-31 shifted by 8
            b.h[0] = *(const v8h*)&Bcol[kb + hs * 8];
            b.h[1] = *(const v8h*)&Bcol[kb + 16 + hs * 8];
            c = __builtin_amdgcn_wmma_f32_16x16x32_f16(false, a.v, false, b.v,
                                                       (short)0, c, false, false);
        }
    } else {
        // K < 32 (layer 0, K=8): only lanes with hs==0 carry non-zero K values.
        frag16 a, b;
        a.v = (v16h)(_Float16)0.0f;
        b.v = (v16h)(_Float16)0.0f;
        if (hs == 0) {
            a.h[0] = *(const v8h*)&Arow[0];
            b.h[0] = *(const v8h*)&Bcol[0];
        }
        c = __builtin_amdgcn_wmma_f32_16x16x32_f16(false, a.v, false, b.v,
                                                   (short)0, c, false, false);
    }

    // C tile store: uniform range check -> fast path is 8 unguarded b32 stores.
    float* __restrict__ Crow = C + (size_t)(m0 + hs * 8) * HID + col;
    if (m0 + 16 <= N) {
#pragma unroll
        for (int r = 0; r < 8; ++r) Crow[(size_t)r * HID] = c[r];
    } else {
#pragma unroll
        for (int r = 0; r < 8; ++r) {
            if (m0 + hs * 8 + r < N) Crow[(size_t)r * HID] = c[r];
        }
    }
}

// ---------- GCN aggregation ----------
__global__ void k_gcn_edge(const float* __restrict__ h, const int* __restrict__ src,
                           const int* __restrict__ dst, const float* __restrict__ dinv,
                           float* __restrict__ out, int E) {
    int e = blockIdx.x * (blockDim.x >> 5) + (threadIdx.x >> 5);
    int lane = threadIdx.x & 31;
    if (e >= E) return;
    int s = src[e], d = dst[e];
    float nrm = dinv[s] * dinv[d];
    const float4 v = *(const float4*)&h[s * HID + lane * 4];
    float* o = &out[d * HID + lane * 4];
    atomicAdd(o + 0, nrm * v.x);
    atomicAdd(o + 1, nrm * v.y);
    atomicAdd(o + 2, nrm * v.z);
    atomicAdd(o + 3, nrm * v.w);
}
__global__ void k_gcn_self_bias(const float* __restrict__ h, const float* __restrict__ dinv,
                                const float* __restrict__ b, float* __restrict__ out, int N) {
    int i = blockIdx.x * blockDim.x + threadIdx.x;
    if (i >= N * HID) return;
    int n = i >> 7, f = i & 127;
    float di = dinv[n];
    out[i] += h[i] * di * di + b[f];
}

// ---------- batchnorm ----------
__global__ __launch_bounds__(128)
void k_bn_stats(const float* __restrict__ x, float* __restrict__ stat, int N) {
    int f = threadIdx.x;                      // 128 features
    float s = 0.0f, q = 0.0f;
    for (int n = blockIdx.x; n < N; n += gridDim.x) {
        float v = x[n * HID + f];
        s += v; q += v * v;
    }
    atomicAdd(&stat[f], s);
    atomicAdd(&stat[HID + f], q);
}
__global__ __launch_bounds__(128)
void k_bn_final(const float* __restrict__ stat, const float* __restrict__ g,
                const float* __restrict__ be, float* __restrict__ prm, float invN) {
    int f = threadIdx.x;
    float mu  = stat[f] * invN;
    float var = stat[HID + f] * invN - mu * mu;
    float sc  = rsqrtf(var + 1e-5f) * g[f];
    prm[f] = sc;
    prm[HID + f] = be[f] - mu * sc;
}
// normalize + ReLU, emit f16 activation (only ever consumed as GEMM A operand)
__global__ void k_bn_relu_f16(const float* __restrict__ x, const float* __restrict__ prm,
                              _Float16* __restrict__ xh, int N) {
    int i = blockIdx.x * blockDim.x + threadIdx.x;
    if (i >= N * HID) return;
    int f = i & 127;
    xh[i] = (_Float16)fmaxf(x[i] * prm[f] + prm[HID + f], 0.0f);
}

// ---------- GAT ----------
__global__ void k_gat_attn(const float* __restrict__ hg, const float* __restrict__ att_s,
                           const float* __restrict__ att_d, float* __restrict__ as,
                           float* __restrict__ ad, int N) {
    int idx = blockIdx.x * blockDim.x + threadIdx.x;   // node*4 + head
    if (idx >= N * HEADS) return;
    int n = idx >> 2, h = idx & 3;
    float s = 0.0f, d = 0.0f;
#pragma unroll 8
    for (int j = 0; j < HDIM; ++j) {
        float v = hg[n * HID + h * HDIM + j];
        s += v * att_s[h * HDIM + j];
        d += v * att_d[h * HDIM + j];
    }
    as[idx] = s; ad[idx] = d;
}
__global__ void k_gat_max(const int* __restrict__ src, const int* __restrict__ dst,
                          const float* __restrict__ as, const float* __restrict__ ad,
                          unsigned* __restrict__ m, int E, int N) {
    int idx = blockIdx.x * blockDim.x + threadIdx.x;   // (E+N)*4
    if (idx >= (E + N) * HEADS) return;
    int e = idx >> 2, h = idx & 3;
    int s, d;
    if (e < E) { s = src[e]; d = dst[e]; } else { s = d = e - E; }
    float v = leaky02(as[s * 4 + h] + ad[d * 4 + h]);
    atomicMax(&m[d * 4 + h], f32_ord(v));
}
__global__ void k_gat_acc(const int* __restrict__ src, const int* __restrict__ dst,
                          const float* __restrict__ as, const float* __restrict__ ad,
                          const unsigned* __restrict__ m, const float* __restrict__ hg,
                          float* __restrict__ denom, float* __restrict__ num, int E, int N) {
    int e = blockIdx.x * (blockDim.x >> 5) + (threadIdx.x >> 5);
    int lane = threadIdx.x & 31;
    if (e >= E + N) return;
    int s, d;
    if (e < E) { s = src[e]; d = dst[e]; } else { s = d = e - E; }
    int h = lane >> 3;                                  // 8 lanes per head
    float v  = leaky02(as[s * 4 + h] + ad[d * 4 + h]);
    float ex = expf(v - ord_f32(m[d * 4 + h]));
    if ((lane & 7) == 0) atomicAdd(&denom[d * 4 + h], ex);
    const float4 hv = *(const float4*)&hg[s * HID + lane * 4];
    float* o = &num[d * HID + lane * 4];
    atomicAdd(o + 0, ex * hv.x);
    atomicAdd(o + 1, ex * hv.y);
    atomicAdd(o + 2, ex * hv.z);
    atomicAdd(o + 3, ex * hv.w);
}
__global__ void k_gat_final(float* __restrict__ x, const float* __restrict__ denom,
                            const float* __restrict__ bg, int N) {
    int i = blockIdx.x * blockDim.x + threadIdx.x;
    if (i >= N * HID) return;
    int n = i >> 7, f = i & 127;
    x[i] = x[i] / denom[n * 4 + (f >> 5)] + bg[f];
}

// ---------- pooling ----------
__global__ void k_pool_node(const float* __restrict__ x, const int* __restrict__ batch,
                            float* __restrict__ cnt, float* __restrict__ psum,
                            unsigned* __restrict__ pmax, int N) {
    int n = blockIdx.x * (blockDim.x >> 5) + (threadIdx.x >> 5);
    int lane = threadIdx.x & 31;
    if (n >= N) return;
    int g = batch[n];
    if (lane == 0) atomicAdd(&cnt[g], 1.0f);
    const float4 v = *(const float4*)&x[n * HID + lane * 4];
    float* ps = &psum[g * HID + lane * 4];
    unsigned* pm = &pmax[g * HID + lane * 4];
    atomicAdd(ps + 0, v.x); atomicMax(pm + 0, f32_ord(v.x));
    atomicAdd(ps + 1, v.y); atomicMax(pm + 1, f32_ord(v.y));
    atomicAdd(ps + 2, v.z); atomicMax(pm + 2, f32_ord(v.z));
    atomicAdd(ps + 3, v.w); atomicMax(pm + 3, f32_ord(v.w));
}
__global__ void k_hcat(const float* __restrict__ cnt, const float* __restrict__ psum,
                       const unsigned* __restrict__ pmax, float* __restrict__ hcat, int G) {
    int idx = blockIdx.x * blockDim.x + threadIdx.x;   // G*256
    if (idx >= G * 256) return;
    int g = idx >> 8, f = idx & 255;
    float c = cnt[g];
    float v;
    if (f < HID) v = psum[g * HID + f] / fmaxf(c, 1.0f);
    else         v = (c > 0.0f) ? ord_f32(pmax[g * HID + (f - HID)]) : 0.0f;
    hcat[idx] = v;
}

// ---------- MLP heads ----------
__global__ __launch_bounds__(128)
void k_mlp(const float* __restrict__ hcat, const float* __restrict__ H1,
           const float* __restrict__ Hb1, const float* __restrict__ H2,
           const float* __restrict__ Hb2, float* __restrict__ out, int G) {
    int g = blockIdx.x, t = blockIdx.y, j = threadIdx.x;
    __shared__ float hc[256];
    __shared__ float red[128];
    hc[j]       = hcat[g * 256 + j];
    hc[128 + j] = hcat[g * 256 + 128 + j];
    __syncthreads();
    const float* W = H1 + (size_t)t * 256 * HID + j;
    float acc = Hb1[t * HID + j];
#pragma unroll 8
    for (int d = 0; d < 256; ++d) acc += hc[d] * W[(size_t)d * HID];
    float z = fmaxf(acc, 0.0f);
    red[j] = z * H2[t * HID + j];
    __syncthreads();
    for (int s = 64; s > 0; s >>= 1) {
        if (j < s) red[j] += red[j + s];
        __syncthreads();
    }
    if (j == 0) out[g * 5 + t] = red[0] + Hb2[t];
}

// =====================================================================
extern "C" void kernel_launch(void* const* d_in, const int* in_sizes, int n_in,
                              void* d_out, int out_size, void* d_ws, size_t ws_size,
                              hipStream_t stream) {
    const float* x        = (const float*)d_in[0];
    const int*   ei       = (const int*)d_in[1];
    const int*   batch    = (const int*)d_in[2];
    const float* W0 = (const float*)d_in[3];
    const float* b0 = (const float*)d_in[4];
    const float* g0 = (const float*)d_in[5];
    const float* be0 = (const float*)d_in[6];
    const float* W1 = (const float*)d_in[7];
    const float* b1 = (const float*)d_in[8];
    const float* g1 = (const float*)d_in[9];
    const float* be1 = (const float*)d_in[10];
    const float* W2 = (const float*)d_in[11];
    const float* b2 = (const float*)d_in[12];
    const float* g2 = (const float*)d_in[13];
    const float* be2 = (const float*)d_in[14];
    const float* Wg = (const float*)d_in[15];
    const float* att_s = (const float*)d_in[16];
    const float* att_d = (const float*)d_in[17];
    const float* bg = (const float*)d_in[18];
    const float* H1 = (const float*)d_in[19];
    const float* Hb1 = (const float*)d_in[20];
    const float* H2 = (const float*)d_in[21];
    const float* Hb2 = (const float*)d_in[22];
    float* out = (float*)d_out;

    const int N = in_sizes[0] / 8;
    const int E = in_sizes[1] / 2;
    const int G = out_size / 5;
    const int* src = ei;
    const int* dst = ei + E;

    // ----- workspace carving -----
    char* w = (char*)d_ws;
    auto alloc = [&](size_t bytes) -> void* {
        void* p = (void*)w;
        w += (bytes + 255) & ~(size_t)255;
        return p;
    };
    float*    dinv   = (float*)alloc((size_t)N * 4);
    float*    bufA   = (float*)alloc((size_t)N * HID * 4);   // GEMM out / agg input (f32)
    float*    bufB   = (float*)alloc((size_t)N * HID * 4);   // agg out / GAT num (f32)
    _Float16* actH   = (_Float16*)alloc((size_t)N * HID * 2);// post-BN activation (f16)
    _Float16* xh     = (_Float16*)alloc((size_t)N * 8 * 2);  // input x (f16)
    float*    asrc   = (float*)alloc((size_t)N * 4 * 4);
    float*    adst   = (float*)alloc((size_t)N * 4 * 4);
    unsigned* mmax   = (unsigned*)alloc((size_t)N * 4 * 4);
    float*    denom  = (float*)alloc((size_t)N * 4 * 4);
    float*    bnstat = (float*)alloc(256 * 4);
    float*    bnprm  = (float*)alloc(256 * 4);
    float*    cnt    = (float*)alloc((size_t)G * 4);
    float*    psum   = (float*)alloc((size_t)G * HID * 4);
    unsigned* pmax   = (unsigned*)alloc((size_t)G * HID * 4);
    float*    hcat   = (float*)alloc((size_t)G * 256 * 4);
    _Float16* W0t    = (_Float16*)alloc((size_t)8 * HID * 2);   // transposed f16 weights
    _Float16* W1t    = (_Float16*)alloc((size_t)HID * HID * 2);
    _Float16* W2t    = (_Float16*)alloc((size_t)HID * HID * 2);
    _Float16* Wgt    = (_Float16*)alloc((size_t)HID * HID * 2);

    const int TB = 256;
    auto cdiv = [](long long a, long long b) { return (int)((a + b - 1) / b); };

    // ----- weights -> f16 transposed; x -> f16 -----
    k_w_to_f16_t<<<cdiv(8 * HID, TB), TB, 0, stream>>>(W0, W0t, 8);
    k_w_to_f16_t<<<cdiv(HID * HID, TB), TB, 0, stream>>>(W1, W1t, HID);
    k_w_to_f16_t<<<cdiv(HID * HID, TB), TB, 0, stream>>>(W2, W2t, HID);
    k_w_to_f16_t<<<cdiv(HID * HID, TB), TB, 0, stream>>>(Wg, Wgt, HID);
    k_f32_to_f16<<<cdiv((long long)N * 8, TB), TB, 0, stream>>>(x, xh, N * 8);

    // ----- degrees (self-loops => init 1.0) -----
    k_fill1<<<cdiv(N, TB), TB, 0, stream>>>(dinv, N);
    k_deg_edges<<<cdiv(E, TB), TB, 0, stream>>>(dst, dinv, E);
    k_rsqrt_inplace<<<cdiv(N, TB), TB, 0, stream>>>(dinv, N);

    const int gemmBlocks = cdiv(N, 16);
    const int edgeWaveBlocks = cdiv(E, TB / 32);
    const float invN = 1.0f / (float)N;

    // ----- 3x GCN + BN + ReLU -----
    const _Float16* Wt[3] = {W0t, W1t, W2t};
    const float* bs[3] = {b0, b1, b2};
    const float* gs[3] = {g0, g1, g2};
    const float* bes[3] = {be0, be1, be2};
    for (int L = 0; L < 3; ++L) {
        if (L == 0)
            k_gemm_wmma<8><<<gemmBlocks, TB, 0, stream>>>(xh, Wt[0], bufA, N);
        else
            k_gemm_wmma<HID><<<gemmBlocks, TB, 0, stream>>>(actH, Wt[L], bufA, N);
        hipMemsetAsync(bufB, 0, (size_t)N * HID * 4, stream);
        k_gcn_edge<<<edgeWaveBlocks, TB, 0, stream>>>(bufA, src, dst, dinv, bufB, E);
        k_gcn_self_bias<<<cdiv((long long)N * HID, TB), TB, 0, stream>>>(bufA, dinv, bs[L], bufB, N);
        hipMemsetAsync(bnstat, 0, 256 * 4, stream);
        k_bn_stats<<<256, 128, 0, stream>>>(bufB, bnstat, N);
        k_bn_final<<<1, 128, 0, stream>>>(bnstat, gs[L], bes[L], bnprm, invN);
        k_bn_relu_f16<<<cdiv((long long)N * HID, TB), TB, 0, stream>>>(bufB, bnprm, actH, N);
    }

    // ----- GAT -----
    k_gemm_wmma<HID><<<gemmBlocks, TB, 0, stream>>>(actH, Wgt, bufA, N);   // hg in bufA (f32)
    k_gat_attn<<<cdiv((long long)N * HEADS, TB), TB, 0, stream>>>(bufA, att_s, att_d, asrc, adst, N);
    hipMemsetAsync(mmax, 0, (size_t)N * 4 * 4, stream);                    // ordered-uint 0 == most negative
    k_gat_max<<<cdiv((long long)(E + N) * HEADS, TB), TB, 0, stream>>>(src, dst, asrc, adst, mmax, E, N);
    hipMemsetAsync(denom, 0, (size_t)N * 4 * 4, stream);
    hipMemsetAsync(bufB, 0, (size_t)N * HID * 4, stream);
    k_gat_acc<<<cdiv(E + N, TB / 32), TB, 0, stream>>>(src, dst, asrc, adst, mmax, bufA, denom, bufB, E, N);
    k_gat_final<<<cdiv((long long)N * HID, TB), TB, 0, stream>>>(bufB, denom, bg, N);

    // ----- pooling -----
    hipMemsetAsync(cnt, 0, (size_t)G * 4, stream);
    hipMemsetAsync(psum, 0, (size_t)G * HID * 4, stream);
    hipMemsetAsync(pmax, 0, (size_t)G * HID * 4, stream);
    k_pool_node<<<cdiv(N, TB / 32), TB, 0, stream>>>(bufB, batch, cnt, psum, pmax, N);
    k_hcat<<<cdiv((long long)G * 256, TB), TB, 0, stream>>>(cnt, psum, pmax, hcat, G);

    // ----- MLP heads -----
    k_mlp<<<dim3(G, 5), 128, 0, stream>>>(hcat, H1, Hb1, H2, Hb2, out, G);

    (void)n_in; (void)ws_size;
}
